// SpatialAttention_25099788877900
// MI455X (gfx1250) — matrix operand order repaired and checked by reading, not compiled, not run on hardware
//
#include <hip/hip_runtime.h>

// Problem constants (reference: B=128, N=512, F=1, T=64)
#define BB 128
#define NB 512
#define TB 64
#define JBLK 64      // columns of output per workgroup
#define BSTR 520     // LDS row stride (bf16 elems) for v_s tile: 520*2=1040B -> 4-bank skew, conflict-free

typedef __attribute__((ext_vector_type(8)))  __bf16 bf16x8;
typedef __attribute__((ext_vector_type(16))) __bf16 bf16x16;
typedef __attribute__((ext_vector_type(8)))  float  v8f;

__device__ __forceinline__ float sigmoidf_(float z) {
    return 1.0f / (1.0f + __expf(-z));
}

// ---------------------------------------------------------------------------
// Kernel 1: s[b,n] = dot(x[b,n,0,:], w1);  u[b,n] = w3 * dot(x[b,n,0,:], w2)
// ---------------------------------------------------------------------------
__global__ __launch_bounds__(256) void k_su(const float* __restrict__ x,
                                            const float* __restrict__ w1,
                                            const float* __restrict__ w2,
                                            const float* __restrict__ w3,
                                            float* __restrict__ s_buf,
                                            float* __restrict__ u_buf) {
    int tid = blockIdx.x * blockDim.x + threadIdx.x;   // 0 .. B*N-1
    const float4* xp  = (const float4*)(x + (size_t)tid * TB);
    const float4* w1p = (const float4*)w1;
    const float4* w2p = (const float4*)w2;
    float d1 = 0.f, d2 = 0.f;
    #pragma unroll
    for (int c = 0; c < TB / 4; ++c) {
        float4 xv = xp[c], a = w1p[c], b = w2p[c];
        d1 += xv.x*a.x + xv.y*a.y + xv.z*a.z + xv.w*a.w;
        d2 += xv.x*b.x + xv.y*b.y + xv.z*b.z + xv.w*b.w;
    }
    s_buf[tid] = d1;
    u_buf[tid] = w3[0] * d2;
}

// ---------------------------------------------------------------------------
// Kernel 2: v_s (f32, row j contiguous in k) -> bf16, same layout
// ---------------------------------------------------------------------------
__global__ __launch_bounds__(256) void k_cvt_v(const float* __restrict__ vs,
                                               __bf16* __restrict__ vsb) {
    int t = blockIdx.x * blockDim.x + threadIdx.x;     // each handles 8 elems
    int idx = t * 8;
    float4 a = *(const float4*)(vs + idx);
    float4 b = *(const float4*)(vs + idx + 4);
    bf16x8 o;
    o[0] = (__bf16)a.x; o[1] = (__bf16)a.y; o[2] = (__bf16)a.z; o[3] = (__bf16)a.w;
    o[4] = (__bf16)b.x; o[5] = (__bf16)b.y; o[6] = (__bf16)b.z; o[7] = (__bf16)b.w;
    *(bf16x8*)(vsb + idx) = o;
}

// ---------------------------------------------------------------------------
// Kernel 3: sig[b,i,k] = sigmoid(s[b,i]*u[b,k] + b_s[i,k])  -> bf16
// (product is rank-1: w3 already folded into u)
// ---------------------------------------------------------------------------
__global__ __launch_bounds__(256) void k_sig(const float* __restrict__ s_buf,
                                             const float* __restrict__ u_buf,
                                             const float* __restrict__ bs,
                                             __bf16* __restrict__ sigb) {
    size_t gid = (size_t)blockIdx.x * blockDim.x + threadIdx.x;
    size_t idx = gid * 8;                  // 8 consecutive k per thread
    int b   = (int)(idx >> 18);            // / (512*512)
    int rem = (int)(idx & 262143);
    int i   = rem >> 9;
    int k   = rem & 511;
    float sv = s_buf[b * NB + i];
    const float4* up = (const float4*)(u_buf + b * NB + k);
    const float4* bp = (const float4*)(bs + i * NB + k);
    float4 u0 = up[0], u1 = up[1], b0 = bp[0], b1 = bp[1];
    float z[8] = { fmaf(sv, u0.x, b0.x), fmaf(sv, u0.y, b0.y),
                   fmaf(sv, u0.z, b0.z), fmaf(sv, u0.w, b0.w),
                   fmaf(sv, u1.x, b1.x), fmaf(sv, u1.y, b1.y),
                   fmaf(sv, u1.z, b1.z), fmaf(sv, u1.w, b1.w) };
    bf16x8 o;
    #pragma unroll
    for (int e = 0; e < 8; ++e) o[e] = (__bf16)sigmoidf_(z[e]);
    *(bf16x8*)(sigb + idx) = o;
}

// ---------------------------------------------------------------------------
// Kernel 4: per (b, j-block): S = sig[b] @ v_s^T via WMMA bf16, then fused
// column-softmax over i (all 512 rows live in this workgroup).
// Block: 512 threads = 16 waves; wave (wm in 0..7, wn in 0..1) owns
// rows [wm*64, wm*64+64) x cols [wn*32, wn*32+32)  (4 x 2 WMMA tiles).
// ---------------------------------------------------------------------------
__global__ __launch_bounds__(512) void k_gemm_softmax(
        const __bf16* __restrict__ sigb, const __bf16* __restrict__ vsb,
        float* __restrict__ out) {
    __shared__ __bf16 Bs[JBLK * BSTR];   // v_s[jb..jb+63, 0..511] bf16, padded
    __shared__ float  red[JBLK * 16];    // per-column partials (8 wm x 2 lane-half)
    __shared__ float  colv[JBLK];        // per-column max, then sum

    const int b   = blockIdx.y;
    const int jb  = blockIdx.x * JBLK;
    const int tid = threadIdx.x;

    // ---- stage B tile into LDS: 64 rows x 512 bf16 (each thread: 128 B) ----
    {
        int jl = tid >> 3, ch = tid & 7;
        const bf16x8* sp = (const bf16x8*)(vsb + (size_t)(jb + jl) * NB + ch * 64);
        bf16x8*       dp = (bf16x8*)(Bs + jl * BSTR + ch * 64);
        #pragma unroll
        for (int t = 0; t < 8; ++t) dp[t] = sp[t];
    }
    __syncthreads();

    const int lane = tid & 31;
    const int wid  = tid >> 5;
    const int wm   = wid & 7;      // i stripe (64 rows)
    const int wn   = wid >> 3;     // j half (32 cols)
    const int r    = lane & 15;
    const int h    = lane >> 4;

    const __bf16* Ab = sigb + (size_t)b * NB * NB;

    const v8f vzero = {0.f,0.f,0.f,0.f,0.f,0.f,0.f,0.f};
    v8f acc[4][2];
    #pragma unroll
    for (int tm = 0; tm < 4; ++tm)
        #pragma unroll
        for (int tn = 0; tn < 2; ++tn) acc[tm][tn] = vzero;

    union AF { bf16x16 v; bf16x8 h2[2]; };

    // ---- K loop: 512 in steps of 32 (v_wmma_f32_16x16x32_bf16) ----
    for (int k0 = 0; k0 < NB; k0 += 32) {
        // A fragments: 16-bit A 16x32 layout: lane r holds row i; lanes 0-15
        // carry K = k0+0..7 & k0+16..23, lanes 16-31 carry K = k0+8..15 & k0+24..31.
        AF a[4];
        #pragma unroll
        for (int tm = 0; tm < 4; ++tm) {
            const __bf16* p = Ab + (size_t)(wm*64 + tm*16 + r) * NB + k0 + h*8;
            a[tm].h2[0] = *(const bf16x8*)p;         // global_load_b128
            a[tm].h2[1] = *(const bf16x8*)(p + 16);  // global_load_b128
        }
        // B fragments: 16-bit B 32x16 layout: lane r holds col j; lanes 0-15
        // carry K = k0+0..7 & k0+16..23, lanes 16-31 carry K = k0+8..15 & k0+24..31.
        AF bb[2];
        #pragma unroll
        for (int tn = 0; tn < 2; ++tn) {
            const __bf16* q = Bs + (wn*32 + tn*16 + r) * BSTR + k0 + h*8;
            bb[tn].h2[0] = *(const bf16x8*)q;        // ds_read_b128
            bb[tn].h2[1] = *(const bf16x8*)(q + 16); // ds_read_b128
        }
        #pragma unroll
        for (int tm = 0; tm < 4; ++tm)
            #pragma unroll
            for (int tn = 0; tn < 2; ++tn)
                acc[tm][tn] = __builtin_amdgcn_wmma_f32_16x16x32_bf16(
                    false, a[tm].v, false, bb[tn].v,
                    (short)0, acc[tm][tn], false, false);
    }

    // ---- fused softmax over i (axis=1). C/D tile layout: element (m,n):
    // n = lane%16, m = vgpr + 8*(lane>>4). Lane owns column c and 32 row vals.
    const int slot = wm * 2 + h;           // 16 partials per column
    #pragma unroll
    for (int tn = 0; tn < 2; ++tn) {
        int c = wn*32 + tn*16 + r;
        float m = -3.0e38f;
        #pragma unroll
        for (int tm = 0; tm < 4; ++tm)
            #pragma unroll
            for (int e = 0; e < 8; ++e) m = fmaxf(m, acc[tm][tn][e]);
        red[c * 16 + slot] = m;
    }
    __syncthreads();
    if (tid < JBLK) {
        float m = -3.0e38f;
        #pragma unroll
        for (int q2 = 0; q2 < 16; ++q2) m = fmaxf(m, red[tid * 16 + q2]);
        colv[tid] = m;
    }
    __syncthreads();
    #pragma unroll
    for (int tn = 0; tn < 2; ++tn) {
        int c = wn*32 + tn*16 + r;
        float m = colv[c];
        float sacc = 0.f;
        #pragma unroll
        for (int tm = 0; tm < 4; ++tm)
            #pragma unroll
            for (int e = 0; e < 8; ++e) {
                float ev = __expf(acc[tm][tn][e] - m);
                acc[tm][tn][e] = ev;
                sacc += ev;
            }
        red[c * 16 + slot] = sacc;
    }
    __syncthreads();
    if (tid < JBLK) {
        float sacc = 0.f;
        #pragma unroll
        for (int q2 = 0; q2 < 16; ++q2) sacc += red[tid * 16 + q2];
        colv[tid] = sacc;
    }
    __syncthreads();
    float* outb = out + (size_t)b * NB * NB + jb;
    #pragma unroll
    for (int tn = 0; tn < 2; ++tn) {
        int c = wn*32 + tn*16 + r;
        float inv = 1.0f / colv[c];
        #pragma unroll
        for (int tm = 0; tm < 4; ++tm)
            #pragma unroll
            for (int e = 0; e < 8; ++e) {
                int i = wm*64 + tm*16 + h*8 + e;
                outb[(size_t)i * NB + c] = acc[tm][tn][e] * inv;
            }
    }
}

// ---------------------------------------------------------------------------
extern "C" void kernel_launch(void* const* d_in, const int* in_sizes, int n_in,
                              void* d_out, int out_size, void* d_ws, size_t ws_size,
                              hipStream_t stream) {
    const float* x  = (const float*)d_in[0];
    const float* w1 = (const float*)d_in[1];
    const float* w2 = (const float*)d_in[2];
    const float* w3 = (const float*)d_in[3];
    const float* bs = (const float*)d_in[4];
    const float* vs = (const float*)d_in[5];
    float* out = (float*)d_out;

    // workspace layout: s (B*N f32) | u (B*N f32) | vsb (N*N bf16) | sigb (B*N*N bf16)
    float*  s_buf = (float*)d_ws;
    float*  u_buf = s_buf + BB * NB;
    __bf16* vsb   = (__bf16*)(u_buf + BB * NB);
    __bf16* sigb  = vsb + NB * NB;

    k_su   <<<(BB * NB) / 256, 256, 0, stream>>>(x, w1, w2, w3, s_buf, u_buf);
    k_cvt_v<<<(NB * NB / 8) / 256, 256, 0, stream>>>(vs, vsb);
    k_sig  <<<(int)(((size_t)BB * NB * NB / 8) / 256), 256, 0, stream>>>(s_buf, u_buf, bs, sigb);

    dim3 grid(NB / JBLK, BB);
    k_gemm_softmax<<<grid, 512, 0, stream>>>(sigb, vsb, out);
}
